// GATLayer_31044023615609
// MI455X (gfx1250) — matrix operand order repaired and checked by reading, not compiled, not run on hardware
//
#include <hip/hip_runtime.h>

// GAT layer (inference) for MI455X / gfx1250, wave32 + WMMA.
// Pipeline: k1: h=x@W (f16 WMMA, f32 accum) -> k2a: s1,s2 -> k2b: pack h into
// f16 B-fragment layout -> k3: single-pass flash masked-softmax-attention with
// V_WMMA_F32_16X16X32_F16, split-K(j) x8 for occupancy, global_prefetch of the
// next adj/hh chunks -> k4: deterministic split combine + 1/l normalize + ELU.

typedef __attribute__((ext_vector_type(16))) _Float16 v16h;
typedef __attribute__((ext_vector_type(8)))  float    v8f;

#define NSZ    8192
#define FIN    128
#define FOUT   64
#define LRELU  0.2f
#define NEGV   (-9e15f)
#define JSPLIT 8
#define TILES  (NSZ / 16)                 // 512 row tiles

// workspace layout (units: float / 4 bytes)
#define WS_H    0                          // h: 8192*64 f32
#define WS_S1   (WS_H + NSZ * FOUT)        // 8192
#define WS_S2   (WS_S1 + NSZ)              // 8192
#define WS_HH   (WS_S2 + NSZ)              // 8192*64 f16 = 262144 u32
#define WS_ACC  (WS_HH + NSZ * FOUT / 2)   // 512*8*1024 f32 partial accumulators
#define WS_ML   (WS_ACC + TILES * JSPLIT * 1024) // 512*8*32 f32 (m,l per row)

// ---------------------------------------------------------------------------
// k1: h = x @ W. One wave per 16-row tile; 16x64 output via 4 kc-chunks x 4 n-tiles.
// A fragment (f16 16x32): lane<16: M=lane, slot s -> K = s;
// lanes>=16 add +16 to K. B fragment (f16 32x16): VGPR v holds rows K=2v,2v+1
// (+16 for lanes>=16), column N = lane&15. C (f32): VGPR v -> M = v + 8*(lane>=16).
__global__ __launch_bounds__(32) void k1_h_gemm(const float* __restrict__ x,
                                                const float* __restrict__ W,
                                                float* __restrict__ h) {
  const int lane = threadIdx.x;
  const int lh   = lane >> 4;
  const int row  = lane & 15;
  const int r0   = blockIdx.x * 16;
  v8f acc[4] = {};
  for (int kc = 0; kc < FIN; kc += 32) {
    v16h af;
#pragma unroll
    for (int s = 0; s < 16; ++s) {
      const int K = (lh ? 16 : 0) + s;
      af[s] = (_Float16)x[(r0 + row) * FIN + kc + K];
    }
#pragma unroll
    for (int nt = 0; nt < 4; ++nt) {
      const int col = nt * 16 + row;
      v16h bf;
#pragma unroll
      for (int v = 0; v < 8; ++v) {
        const int kr = kc + (lh ? 16 : 0) + 2 * v;
        bf[2 * v]     = (_Float16)W[kr * FOUT + col];
        bf[2 * v + 1] = (_Float16)W[(kr + 1) * FOUT + col];
      }
      acc[nt] = __builtin_amdgcn_wmma_f32_16x16x32_f16(
          false, af, false, bf, (short)0, acc[nt], false, false);
    }
  }
#pragma unroll
  for (int nt = 0; nt < 4; ++nt)
#pragma unroll
    for (int v = 0; v < 8; ++v)
      h[(r0 + v + 8 * lh) * FOUT + nt * 16 + row] = acc[nt][v];
}

// ---------------------------------------------------------------------------
// k2a: s1[i] = h[i].a1, s2[i] = h[i].a2  (a = [a1(64); a2(64)])
__global__ void k2_scores(const float* __restrict__ h, const float* __restrict__ a,
                          float* __restrict__ s1, float* __restrict__ s2) {
  const int r = blockIdx.x * blockDim.x + threadIdx.x;
  if (r >= NSZ) return;
  float acc1 = 0.f, acc2 = 0.f;
#pragma unroll 8
  for (int f = 0; f < FOUT; ++f) {
    const float hv = h[r * FOUT + f];
    acc1 += hv * a[f];
    acc2 += hv * a[FOUT + f];
  }
  s1[r] = acc1;
  s2[r] = acc2;
}

// ---------------------------------------------------------------------------
// k2b: pack h into f16 B-fragment-native layout:
// hh[(((jc*4+nt)*32)+lane)*8 + v] = pack_f16(h[j0][col], h[j0+1][col]),
// j0 = jc*32 + 8*(lane>=16) + (v<4 ? 2v : 2v+8), col = nt*16 + (lane&15).
__global__ void k2_pack(const float* __restrict__ h, unsigned int* __restrict__ hh) {
  const int tid  = blockIdx.x * blockDim.x + threadIdx.x;   // 262144 threads
  const int v    = tid & 7;
  const int lane = (tid >> 3) & 31;
  const int nt   = (tid >> 8) & 3;
  const int jc   = tid >> 10;
  const int j0   = jc * 32 + ((lane & 16) ? 8 : 0) + (v < 4 ? 2 * v : 2 * v + 8);
  const int col  = nt * 16 + (lane & 15);
  union { unsigned int u; _Float16 f[2]; } p;
  p.f[0] = (_Float16)h[j0 * FOUT + col];
  p.f[1] = (_Float16)h[(j0 + 1) * FOUT + col];
  hh[tid] = p.u;
}

// ---------------------------------------------------------------------------
// k3: flash-style masked-softmax attention, one wave per (16-row tile, j-split).
// Streams adj exactly once; online max/sum; P in f16 A-fragment slots; 4 WMMA
// per 32-column chunk accumulate the 16x64 unnormalized output.
__global__ __launch_bounds__(32) void k3_flash(const float* __restrict__ adj,
                                               const float* __restrict__ s1w,
                                               const float* __restrict__ s2w,
                                               const unsigned int* __restrict__ hh,
                                               float* __restrict__ accws,
                                               float* __restrict__ mlws) {
  const int lane  = threadIdx.x;
  const int lh    = lane >> 4;
  const int row   = lane & 15;
  const int tile  = blockIdx.x / JSPLIT;
  const int split = blockIdx.x % JSPLIT;
  const float s1r = s1w[tile * 16 + row];
  const float* arow = adj + (size_t)(tile * 16 + row) * NSZ;

  float m = -INFINITY, l = 0.f;
  v8f acc[4] = {};

  const int JSEG   = NSZ / JSPLIT;               // 1024 columns per split
  const int CHUNKS = JSEG / 32;                  // 32 chunks of 32 columns
  const int PF     = 2;                          // prefetch distance (chunks)
  for (int jc = 0; jc < CHUNKS; ++jc) {
    const int j0   = split * JSEG + jc * 32;
    const int base = j0 + (lh ? 8 : 0);

    // Prefetch chunk jc+PF: one instruction covers the whole next 16x32 adj
    // tile (each lane -> 64B of one row), one covers the next 4KB hh block.
    if (jc + PF < CHUNKS) {
      const int jp = j0 + PF * 32;
      __builtin_prefetch(arow + jp + lh * 16, 0, 3);
      __builtin_prefetch(hh + ((size_t)(jp >> 5) * 4 * 32) * 8 + lane * 32, 0, 3);
    }

    // adj / s2 values at this lane's 16 A-fragment K slots (compiler merges to b128)
    float av[16], sv[16];
#pragma unroll
    for (int t = 0; t < 8; ++t) {
      av[t]     = arow[base + t];       sv[t]     = s2w[base + t];
      av[8 + t] = arow[base + 16 + t];  sv[8 + t] = s2w[base + 16 + t];
    }

    float e[16];
#pragma unroll
    for (int s = 0; s < 16; ++s) {
      const float pre = s1r + sv[s];
      const float lr  = pre >= 0.f ? pre : LRELU * pre;
      e[s] = av[s] > 0.f ? av[s] * lr : NEGV;    // reference masking semantics
    }

    // online softmax: chunk max (16 local + partner lane via swizzle)
    float mc = e[0];
#pragma unroll
    for (int s = 1; s < 16; ++s) mc = fmaxf(mc, e[s]);
    mc = fmaxf(mc, __shfl_xor(mc, 16, 32));
    const float newm  = fmaxf(m, mc);
    const float alpha = __expf(m - newm);        // first chunk: exp(-inf)=0
    m = newm;

    float rs = 0.f;
    v16h af;
#pragma unroll
    for (int s = 0; s < 16; ++s) {
      const float p = __expf(e[s] - m);
      rs += p;
      af[s] = (_Float16)p;
    }
    rs += __shfl_xor(rs, 16, 32);
    l = l * alpha + rs;

    // rescale accumulators: C element (v,lane) belongs to row v + 8*lh
    float al[8];
#pragma unroll
    for (int v = 0; v < 8; ++v) al[v] = __shfl(alpha, v + 8 * lh, 32);
#pragma unroll
    for (int nt = 0; nt < 4; ++nt)
#pragma unroll
      for (int v = 0; v < 8; ++v) acc[nt][v] *= al[v];

    const int jcg = j0 >> 5;
#pragma unroll
    for (int nt = 0; nt < 4; ++nt) {
      const v16h bf = *(const v16h*)(hh + ((size_t)(jcg * 4 + nt) * 32 + lane) * 8);
      acc[nt] = __builtin_amdgcn_wmma_f32_16x16x32_f16(
          false, af, false, bf, (short)0, acc[nt], false, false);
    }
  }

  const size_t ob = (size_t)blockIdx.x * 1024;
#pragma unroll
  for (int nt = 0; nt < 4; ++nt)
#pragma unroll
    for (int v = 0; v < 8; ++v)
      accws[ob + nt * 256 + v * 32 + lane] = acc[nt][v];
  if (lane < 16) {                                // lanes m and m+16 agree
    mlws[blockIdx.x * 32 + lane * 2]     = m;
    mlws[blockIdx.x * 32 + lane * 2 + 1] = l;
  }
}

// ---------------------------------------------------------------------------
// k4: deterministic combine of JSPLIT partials, normalize, ELU. 1 thread/output.
__global__ void k4_combine(const float* __restrict__ accws,
                           const float* __restrict__ mlws,
                           float* __restrict__ out) {
  const int idx = blockIdx.x * blockDim.x + threadIdx.x;
  if (idx >= NSZ * FOUT) return;
  const int r = idx >> 6, f = idx & 63;
  const int tile = r >> 4, mr = r & 15;

  float ms[JSPLIT], ls[JSPLIT];
  float M = -INFINITY;
#pragma unroll
  for (int s = 0; s < JSPLIT; ++s) {
    ms[s] = mlws[(tile * JSPLIT + s) * 32 + mr * 2];
    ls[s] = mlws[(tile * JSPLIT + s) * 32 + mr * 2 + 1];
    M = fmaxf(M, ms[s]);
  }
  const int nt = f >> 4, colIn = f & 15;
  const int v = mr & 7;
  const int lane = colIn + ((mr >= 8) ? 16 : 0);
  float L = 0.f, S = 0.f;
#pragma unroll
  for (int s = 0; s < JSPLIT; ++s) {
    const float w = __expf(ms[s] - M);
    L += ls[s] * w;
    S += accws[(size_t)(tile * JSPLIT + s) * 1024 + nt * 256 + v * 32 + lane] * w;
  }
  const float hp = S / L;
  out[idx] = hp > 0.f ? hp : (__expf(hp) - 1.f);  // ELU
}

// ---------------------------------------------------------------------------
extern "C" void kernel_launch(void* const* d_in, const int* in_sizes, int n_in,
                              void* d_out, int out_size, void* d_ws, size_t ws_size,
                              hipStream_t stream) {
  const float* x   = (const float*)d_in[0];
  const float* adj = (const float*)d_in[1];
  const float* W   = (const float*)d_in[2];
  const float* a   = (const float*)d_in[3];
  float* ws = (float*)d_ws;
  float* h          = ws + WS_H;
  float* s1         = ws + WS_S1;
  float* s2         = ws + WS_S2;
  unsigned int* hh  = (unsigned int*)(ws + WS_HH);
  float* accws      = ws + WS_ACC;
  float* mlws       = ws + WS_ML;
  float* out        = (float*)d_out;

  k1_h_gemm<<<TILES, 32, 0, stream>>>(x, W, h);
  k2_scores<<<NSZ / 256, 256, 0, stream>>>(h, a, s1, s2);
  k2_pack<<<(NSZ * FOUT / 2) / 256, 256, 0, stream>>>(h, hh);
  k3_flash<<<TILES * JSPLIT, 32, 0, stream>>>(adj, s1, s2, hh, accws, mlws);
  k4_combine<<<(NSZ * FOUT) / 256, 256, 0, stream>>>(accws, mlws, out);
}
